// HeteroTransportCell_43885975830665
// MI455X (gfx1250) — matrix operand order repaired and checked by reading, not compiled, not run on hardware
//
#include <hip/hip_runtime.h>

// ---------------------------------------------------------------------------
// Types & WMMA helpers (CDNA5 / gfx1250, wave32, V_WMMA_F32_16X16X32_F16)
// ---------------------------------------------------------------------------
typedef __attribute__((ext_vector_type(16))) _Float16 v16h;
typedef __attribute__((ext_vector_type(8)))  _Float16 v8h;
typedef __attribute__((ext_vector_type(8)))  float    v8f;

union H16 { v16h v; v8h h[2]; };

#define WMMA_F16(a, b, c) \
  __builtin_amdgcn_wmma_f32_16x16x32_f16(false, (a), false, (b), (short)0, (c), false, false)

// A fragment (16x32 f16, MxK): lane l holds row M=l&15; hi=l>>4 selects K half.
// halfs i<8  -> K = kc + hi*8 + i
// halfs i>=8 -> K = kc + 16 + hi*8 + (i-8)
__device__ inline v16h load_a_frag(const _Float16* A, int lda, int row0, int kc, int lane) {
  int row = row0 + (lane & 15);
  int hi  = (lane >> 4) & 1;
  const _Float16* p = A + (size_t)row * lda + kc + hi * 8;
  H16 u;
  u.h[0] = *(const v8h*)(p);
  u.h[1] = *(const v8h*)(p + 16);
  return u.v;
}

// B fragment (32x16 f16, KxN), pre-packed per (ktile,ntile): lane-major, 16 halfs/lane.
__device__ inline v16h load_b_frag(const _Float16* Bp, int ntiles, int kt, int nt, int lane) {
  const _Float16* p = Bp + (((size_t)(kt * ntiles + nt) * 32 + lane) * 16);
  H16 u;
  u.h[0] = *(const v8h*)(p);
  u.h[1] = *(const v8h*)(p + 8);
  return u.v;
}

__device__ inline float wave_sum(float v) {
  for (int o = 16; o > 0; o >>= 1) v += __shfl_xor(v, o, 32);
  return v;
}

__device__ inline float sigm(float x) { return 1.f / (1.f + expf(-x)); }

__device__ inline void atomicMaxF(float* addr, float v) {
  if (v >= 0.f) atomicMax((int*)addr, __float_as_int(v));
  else          atomicMin((unsigned int*)addr, __float_as_uint(v));
}

// ---------------------------------------------------------------------------
// Prep kernels
// ---------------------------------------------------------------------------
__global__ void f32_to_f16_k(const float* __restrict__ s, _Float16* __restrict__ d, int n) {
  int i = blockIdx.x * 256 + threadIdx.x;
  if (i < n) d[i] = (_Float16)s[i];
}

__global__ void fill_f32_k(float* __restrict__ p, float v, int n) {
  int i = blockIdx.x * 256 + threadIdx.x;
  if (i < n) p[i] = v;
}

// Pack W[K x N] (f32 row-major) into B-fragment order (f16), zero-padding K to Kpad.
__global__ void pack_b_k(const float* __restrict__ W, _Float16* __restrict__ dst,
                         int K, int N, int Kpad) {
  int idx = blockIdx.x * 256 + threadIdx.x;
  int ntiles = N >> 4;
  int total = (Kpad >> 5) * ntiles * 512;
  if (idx >= total) return;
  int i    = idx & 15;
  int lane = (idx >> 4) & 31;
  int t    = idx >> 9;
  int kt = t / ntiles, nt = t - kt * ntiles;
  int n  = nt * 16 + (lane & 15);
  int hi = lane >> 4;
  int k  = kt * 32 + hi * 8 + (i & 7) + (i >> 3) * 16;
  float v = (k < K) ? W[(size_t)k * N + n] : 0.f;
  dst[idx] = (_Float16)v;
}

// Fold ese second layer into a vector:  wv = w2 @ bw ,  c0 = b2.bw + bb
__global__ void fold_bw_k(const float* __restrict__ w2, const float* __restrict__ b2,
                          const float* __restrict__ bw, const float* __restrict__ bb,
                          float* __restrict__ wv, float* __restrict__ c0) {
  int k = threadIdx.x;  // 128 threads
  float acc = 0.f;
  for (int o = 0; o < 128; ++o) acc += w2[k * 128 + o] * bw[o];
  wv[k] = acc;
  if (k == 0) {
    float c = bb[0];
    for (int o = 0; o < 128; ++o) c += b2[o] * bw[o];
    c0[0] = c;
  }
}

// ---------------------------------------------------------------------------
// Generic dense WMMA GEMM over node rows:
//   out = [relu]( A[rows x K](f16) @ Bpacked + bias ) [+ addsrc]
// rows % 16 == 0, K % 32 == 0, N % 16 == 0.  One wave per 16-row strip.
// ---------------------------------------------------------------------------
__global__ __launch_bounds__(256) void gemm_wmma_k(
    const _Float16* __restrict__ A, int lda, const _Float16* __restrict__ Bp,
    const float* __restrict__ bias, const float* __restrict__ addsrc,
    float* __restrict__ outf, _Float16* __restrict__ outh,
    int rows, int K, int N, int relu) {
  int wave = (blockIdx.x * blockDim.x + threadIdx.x) >> 5;
  int lane = threadIdx.x & 31;
  int rowTiles = rows >> 4;
  if (wave >= rowTiles) return;
  int r0 = wave << 4;
  int ntiles = N >> 4, ktiles = K >> 5;
  for (int nt = 0; nt < ntiles; ++nt) {
    v8f c = {};
    for (int kt = 0; kt < ktiles; ++kt) {
      v16h a = load_a_frag(A, lda, r0, kt * 32, lane);
      v16h b = load_b_frag(Bp, ntiles, kt, nt, lane);
      c = WMMA_F16(a, b, c);
    }
    int col = nt * 16 + (lane & 15);
    int hi  = lane >> 4;
    for (int r = 0; r < 8; ++r) {
      int row = r0 + r + 8 * hi;
      float v = c[r];
      if (bias)   v += bias[col];
      if (relu)   v = fmaxf(v, 0.f);
      if (addsrc) v += addsrc[(size_t)row * N + col];
      if (outf) outf[(size_t)row * N + col] = v;
      if (outh) outh[(size_t)row * N + col] = (_Float16)v;
    }
  }
}

// ---------------------------------------------------------------------------
// StaticDynamicEdgeMP: one wave per 16-edge tile; WMMA MLPs from LDS-gathered
// tiles, then b*g-scaled payload scatter-added into msg[dst].
// Gathers of h rows done in 16-byte v8h chunks (rows are 96 halfs, 16B-aligned).
// ---------------------------------------------------------------------------
__global__ __launch_bounds__(128) void sd_edge_k(
    const int* __restrict__ ei, const float* __restrict__ ea,
    const float* __restrict__ xs, const _Float16* __restrict__ h16,
    const _Float16* __restrict__ eseW1p, const float* __restrict__ ese_b1,
    const float* __restrict__ wv, const float* __restrict__ c0,
    const _Float16* __restrict__ gateW1p, const float* __restrict__ gate_b1,
    const float* __restrict__ gate_w2, const float* __restrict__ gate_b2,
    const _Float16* __restrict__ payW1p, const float* __restrict__ pay_b1,
    const _Float16* __restrict__ payW2p, const float* __restrict__ pay_b2,
    float* __restrict__ msg, int E) {
  __shared__ __align__(16) _Float16 At[4][16 * 192];
  __shared__ __align__(16) _Float16 Ht[4][16 * 128];
  __shared__ int   sidx[4][16];
  __shared__ int   didx[4][16];
  __shared__ float bsc[4][16];
  __shared__ float bg[4][16];

  const int w    = threadIdx.x >> 5;
  const int lane = threadIdx.x & 31;
  const int e0   = (blockIdx.x * 4 + w) * 16;

  if (lane < 16) {
    sidx[w][lane] = ei[e0 + lane];
    didx[w][lane] = ei[E + e0 + lane];
  }
  __syncthreads();

  // ---- phase 1: ese layer-1 input (K=32 padded: ea|xs_src|xs_dst|0)
  for (int p = lane; p < 16 * 32; p += 32) {
    int r = p >> 5, c = p & 31, e = e0 + r;
    float v;
    if (c < 8)       v = ea[(size_t)e * 8 + c];
    else if (c < 14) v = xs[(size_t)sidx[w][r] * 6 + (c - 8)];
    else if (c < 20) v = xs[(size_t)didx[w][r] * 6 + (c - 14)];
    else             v = 0.f;
    At[w][p] = (_Float16)v;
  }
  __syncthreads();
  for (int nt = 0; nt < 8; ++nt) {
    v8f c = {};
    v16h a = load_a_frag(&At[w][0], 32, 0, 0, lane);
    v16h b = load_b_frag(eseW1p, 8, 0, nt, lane);
    c = WMMA_F16(a, b, c);
    int col = nt * 16 + (lane & 15), hi = lane >> 4;
    for (int r = 0; r < 8; ++r)
      Ht[w][(r + 8 * hi) * 128 + col] = (_Float16)fmaxf(c[r] + ese_b1[col], 0.f);
  }
  __syncthreads();
  {
    // b = softplus(hidden . wv + c0): 32 lanes, half-row each, pair-combine.
    int r = lane & 15, hp = lane >> 4;
    float acc = 0.f;
    for (int k = 0; k < 64; ++k)
      acc += (float)Ht[w][r * 128 + hp * 64 + k] * wv[hp * 64 + k];
    acc += __shfl_xor(acc, 16, 32);
    if (lane < 16) {
      float x = acc + c0[0];
      bsc[w][r] = (x > 20.f) ? x : log1pf(expf(x));
    }
  }
  __syncthreads();

  // ---- phase 2: gate MLP input (K=192): [h_src | h_dst], v8h chunked gather
  for (int q = lane; q < 16 * 24; q += 32) {  // 24 chunks of 8 halfs per row
    int r = q / 24, cc = (q - r * 24) * 8;
    const _Float16* src = (cc < 96)
        ? h16 + (size_t)sidx[w][r] * 96 + cc
        : h16 + (size_t)didx[w][r] * 96 + (cc - 96);
    *(v8h*)&At[w][r * 192 + cc] = *(const v8h*)src;
  }
  __syncthreads();
  for (int nt = 0; nt < 8; ++nt) {
    v8f c = {};
    for (int kt = 0; kt < 6; ++kt) {
      v16h a = load_a_frag(&At[w][0], 192, 0, kt * 32, lane);
      v16h b = load_b_frag(gateW1p, 8, kt, nt, lane);
      c = WMMA_F16(a, b, c);
    }
    int col = nt * 16 + (lane & 15), hi = lane >> 4;
    for (int r = 0; r < 8; ++r)
      Ht[w][(r + 8 * hi) * 128 + col] = (_Float16)fmaxf(c[r] + gate_b1[col], 0.f);
  }
  __syncthreads();
  {
    int r = lane & 15, hp = lane >> 4;
    float acc = 0.f;
    for (int k = 0; k < 64; ++k)
      acc += (float)Ht[w][r * 128 + hp * 64 + k] * gate_w2[hp * 64 + k];
    acc += __shfl_xor(acc, 16, 32);
    if (lane < 16)
      bg[w][r] = bsc[w][r] * sigm(acc + gate_b2[0]);
  }
  __syncthreads();

  // ---- phase 3: payload MLP; A reuses cols 0..95 of gate tile (h_src, lda=192)
  for (int nt = 0; nt < 8; ++nt) {
    v8f c = {};
    for (int kt = 0; kt < 3; ++kt) {
      v16h a = load_a_frag(&At[w][0], 192, 0, kt * 32, lane);
      v16h b = load_b_frag(payW1p, 8, kt, nt, lane);
      c = WMMA_F16(a, b, c);
    }
    int col = nt * 16 + (lane & 15), hi = lane >> 4;
    for (int r = 0; r < 8; ++r)
      Ht[w][(r + 8 * hi) * 128 + col] = (_Float16)fmaxf(c[r] + pay_b1[col], 0.f);
  }
  __syncthreads();
  for (int nt = 0; nt < 4; ++nt) {
    v8f c = {};
    for (int kt = 0; kt < 4; ++kt) {
      v16h a = load_a_frag(&Ht[w][0], 128, 0, kt * 32, lane);
      v16h b = load_b_frag(payW2p, 4, kt, nt, lane);
      c = WMMA_F16(a, b, c);
    }
    int col = nt * 16 + (lane & 15), hi = lane >> 4;
    for (int r = 0; r < 8; ++r) {
      int row = r + 8 * hi;
      float v = (c[r] + pay_b2[col]) * bg[w][row];
      atomicAdd(&msg[(size_t)didx[w][row] * 64 + col], v);
    }
  }
}

// ---------------------------------------------------------------------------
// GATv2 edge passes (segment softmax over destinations)
// ---------------------------------------------------------------------------
__global__ void gat_logits_k(const int* __restrict__ ei, const float* __restrict__ xl,
                             const float* __restrict__ xr, const float* __restrict__ att,
                             float* __restrict__ est, float* __restrict__ mbuf, int E) {
  int t = blockIdx.x * 256 + threadIdx.x;
  if (t >= E * 4) return;
  int e = t >> 2, hh = t & 3;
  int s = ei[e], d = ei[E + e];
  float acc = 0.f;
  for (int c = 0; c < 16; ++c) {
    float x = xl[(size_t)s * 64 + hh * 16 + c] + xr[(size_t)d * 64 + hh * 16 + c];
    x = (x > 0.f) ? x : 0.2f * x;  // leaky_relu 0.2
    acc += x * att[hh * 16 + c];
  }
  est[t] = acc;
  atomicMaxF(&mbuf[(size_t)d * 4 + hh], acc);
}

__global__ void gat_exp_k(const int* __restrict__ ei, float* __restrict__ est,
                          const float* __restrict__ mbuf, float* __restrict__ ssum, int E) {
  int t = blockIdx.x * 256 + threadIdx.x;
  if (t >= E * 4) return;
  int e = t >> 2, hh = t & 3;
  int d = ei[E + e];
  float ex = expf(est[t] - mbuf[(size_t)d * 4 + hh]);
  est[t] = ex;
  atomicAdd(&ssum[(size_t)d * 4 + hh], ex);
}

__global__ void gat_scatter_k(const int* __restrict__ ei, const float* __restrict__ est,
                              const float* __restrict__ ssum, const float* __restrict__ xl,
                              float* __restrict__ aout, int E) {
  int t = blockIdx.x * 256 + threadIdx.x;
  if (t >= E * 4) return;
  int e = t >> 2, hh = t & 3;
  int s = ei[e], d = ei[E + e];
  float a = est[t] / (ssum[(size_t)d * 4 + hh] + 1e-16f);
  for (int c = 0; c < 16; ++c)
    atomicAdd(&aout[(size_t)d * 64 + hh * 16 + c], a * xl[(size_t)s * 64 + hh * 16 + c]);
}

// ---------------------------------------------------------------------------
// Node update: dyn-proj + LN(dyn) + LN(msg) -> xcat(f16);  GRU + LN -> out
// One wave per node; wave32 shuffle reductions.
// ---------------------------------------------------------------------------
__global__ __launch_bounds__(256) void node_pre_k(
    const float* __restrict__ xdyn, const float* __restrict__ dynw,
    const float* __restrict__ dynb, const float* __restrict__ dlng,
    const float* __restrict__ dlnb, const float* __restrict__ msg,
    const float* __restrict__ mlng, const float* __restrict__ mlnb,
    _Float16* __restrict__ xcat, int n) {
  int node = (blockIdx.x * blockDim.x + threadIdx.x) >> 5;
  int lane = threadIdx.x & 31;
  if (node >= n) return;
  int c0 = lane, c1 = lane + 32;
  float d0 = dynb[c0], d1 = dynb[c1];
  for (int k = 0; k < 4; ++k) {
    float x = xdyn[(size_t)node * 4 + k];
    d0 += x * dynw[k * 64 + c0];
    d1 += x * dynw[k * 64 + c1];
  }
  float mu = wave_sum(d0 + d1) * (1.f / 64.f);
  float va = wave_sum((d0 - mu) * (d0 - mu) + (d1 - mu) * (d1 - mu)) * (1.f / 64.f);
  float rs = rsqrtf(va + 1e-5f);
  xcat[(size_t)node * 128 + c0] = (_Float16)((d0 - mu) * rs * dlng[c0] + dlnb[c0]);
  xcat[(size_t)node * 128 + c1] = (_Float16)((d1 - mu) * rs * dlng[c1] + dlnb[c1]);

  float m0 = msg[(size_t)node * 64 + c0], m1 = msg[(size_t)node * 64 + c1];
  mu = wave_sum(m0 + m1) * (1.f / 64.f);
  va = wave_sum((m0 - mu) * (m0 - mu) + (m1 - mu) * (m1 - mu)) * (1.f / 64.f);
  rs = rsqrtf(va + 1e-5f);
  xcat[(size_t)node * 128 + 64 + c0] = (_Float16)((m0 - mu) * rs * mlng[c0] + mlnb[c0]);
  xcat[(size_t)node * 128 + 64 + c1] = (_Float16)((m1 - mu) * rs * mlng[c1] + mlnb[c1]);
}

__global__ __launch_bounds__(256) void gru_ln_k(
    const float* __restrict__ gi, const float* __restrict__ gh,
    const float* __restrict__ h, const float* __restrict__ lng,
    const float* __restrict__ lnb, float* __restrict__ out, int n) {
  int node = (blockIdx.x * blockDim.x + threadIdx.x) >> 5;
  int lane = threadIdx.x & 31;
  if (node >= n) return;
  float hr[3];
  for (int j = 0; j < 3; ++j) {
    int c = lane + 32 * j;
    size_t b = (size_t)node * 288 + c;
    float r  = sigm(gi[b] + gh[b]);
    float z  = sigm(gi[b + 96] + gh[b + 96]);
    float nn = tanhf(gi[b + 192] + r * gh[b + 192]);
    hr[j] = (1.f - z) * nn + z * h[(size_t)node * 96 + c];
  }
  float mu = wave_sum(hr[0] + hr[1] + hr[2]) * (1.f / 96.f);
  float va = wave_sum((hr[0] - mu) * (hr[0] - mu) + (hr[1] - mu) * (hr[1] - mu) +
                      (hr[2] - mu) * (hr[2] - mu)) * (1.f / 96.f);
  float rs = rsqrtf(va + 1e-5f);
  for (int j = 0; j < 3; ++j) {
    int c = lane + 32 * j;
    out[(size_t)node * 96 + c] = (hr[j] - mu) * rs * lng[c] + lnb[c];
  }
}

// ---------------------------------------------------------------------------
// Host orchestration
// ---------------------------------------------------------------------------
extern "C" void kernel_launch(void* const* d_in, const int* in_sizes, int n_in,
                              void* d_out, int out_size, void* d_ws, size_t ws_size,
                              hipStream_t stream) {
  (void)in_sizes; (void)n_in; (void)out_size; (void)ws_size;
  constexpr int NN = 50000, EFn = 400000, ECn = 200000;

  // Input indices (jax tree-flatten, dict keys sorted recursively)
  enum { IN_XS1 = 0, IN_XS2, IN_XD1, IN_XD2, IN_H1, IN_H2, IN_EA1, IN_EA2,
         IN_EIF1, IN_EIF2, IN_EI12, IN_EI21 };
  constexpr int PF1 = 12, PF2 = 26, PG12 = 40, PG21 = 51, PN1 = 62, PN2 = 74;
  // flow offsets: bb,bw,ese{b1,b2,w1,w2},gate{b1,b2,w1,w2},pay{b1,b2,w1,w2}
  enum { F_BB = 0, F_BW, F_ESE_B1, F_ESE_B2, F_ESE_W1, F_ESE_W2,
         F_GATE_B1, F_GATE_B2, F_GATE_W1, F_GATE_W2,
         F_PAY_B1, F_PAY_B2, F_PAY_W1, F_PAY_W2 };
  // gat offsets: att,bias,bl,br,ffn{b1,b2,w1,w2},wl,wr,wres
  enum { G_ATT = 0, G_BIAS, G_BL, G_BR, G_FFN_B1, G_FFN_B2, G_FFN_W1, G_FFN_W2,
         G_WL, G_WR, G_WRES };
  // node offsets: bhh,bih,dyn_b,dyn_ln{b,g},dyn_w,h_ln{b,g},msg_ln{b,g},whh,wih
  enum { ND_BHH = 0, ND_BIH, ND_DYN_B, ND_DLN_B, ND_DLN_G, ND_DYN_W,
         ND_HLN_B, ND_HLN_G, ND_MLN_B, ND_MLN_G, ND_WHH, ND_WIH };

  auto F = [&](int i) { return (const float*)d_in[i]; };
  auto I = [&](int i) { return (const int*)d_in[i]; };

  char* base = (char*)d_ws;
  size_t off = 0;
  auto alloc = [&](size_t bytes) -> char* {
    char* p = base + off;
    off = (off + bytes + 255) & ~(size_t)255;
    return p;
  };

  _Float16* h1h = (_Float16*)alloc((size_t)NN * 96 * 2);
  _Float16* h2h = (_Float16*)alloc((size_t)NN * 96 * 2);

  struct FP { _Float16 *w1p, *gw1p, *pw1p, *pw2p; float *wv, *c0; } fp[2];
  for (int f = 0; f < 2; ++f) {
    fp[f].w1p  = (_Float16*)alloc(8  * 512 * 2);
    fp[f].gw1p = (_Float16*)alloc(48 * 512 * 2);
    fp[f].pw1p = (_Float16*)alloc(24 * 512 * 2);
    fp[f].pw2p = (_Float16*)alloc(16 * 512 * 2);
    fp[f].wv   = (float*)alloc(128 * 4);
    fp[f].c0   = (float*)alloc(4);
  }
  struct GP { _Float16 *wlp, *wrp, *wresp, *f1p, *f2p; } gp[2];  // 0=g12, 1=g21
  for (int g = 0; g < 2; ++g) {
    gp[g].wlp   = (_Float16*)alloc(12 * 512 * 2);
    gp[g].wrp   = (_Float16*)alloc(12 * 512 * 2);
    gp[g].wresp = (_Float16*)alloc(12 * 512 * 2);
    gp[g].f1p   = (_Float16*)alloc(16 * 512 * 2);
    gp[g].f2p   = (_Float16*)alloc(16 * 512 * 2);
  }
  struct NP { _Float16 *wihp, *whhp; } np[2];
  for (int t = 0; t < 2; ++t) {
    np[t].wihp = (_Float16*)alloc(72 * 512 * 2);
    np[t].whhp = (_Float16*)alloc(54 * 512 * 2);
  }

  float* msg1 = (float*)alloc((size_t)NN * 64 * 4);
  float* msg2 = (float*)alloc((size_t)NN * 64 * 4);
  float* xl   = (float*)alloc((size_t)NN * 64 * 4);
  float* xr   = (float*)alloc((size_t)NN * 64 * 4);
  float* est  = (float*)alloc((size_t)ECn * 4 * 4);
  float* mbuf = (float*)alloc((size_t)NN * 4 * 4);
  float* ssum = (float*)alloc((size_t)NN * 4 * 4);
  float* aout = (float*)alloc((size_t)NN * 64 * 4);
  _Float16* attin = (_Float16*)alloc((size_t)NN * 64 * 2);
  _Float16* ffnh  = (_Float16*)alloc((size_t)NN * 128 * 2);
  _Float16* xcat  = (_Float16*)alloc((size_t)NN * 128 * 2);
  float* gi = (float*)alloc((size_t)NN * 288 * 4);
  float* gh = (float*)alloc((size_t)NN * 288 * 4);

  auto fill = [&](float* p, float v, int n) {
    fill_f32_k<<<(n + 255) / 256, 256, 0, stream>>>(p, v, n);
  };
  auto packB = [&](const float* W, _Float16* dst, int K, int N, int Kpad) {
    int total = (Kpad / 32) * (N / 16) * 512;
    pack_b_k<<<(total + 255) / 256, 256, 0, stream>>>(W, dst, K, N, Kpad);
  };
  auto gemm = [&](const _Float16* A, int lda, const _Float16* Bp, const float* bias,
                  const float* addsrc, float* outf, _Float16* outh,
                  int rows, int K, int N, int relu) {
    int threads = (rows / 16) * 32;
    gemm_wmma_k<<<(threads + 255) / 256, 256, 0, stream>>>(
        A, lda, Bp, bias, addsrc, outf, outh, rows, K, N, relu);
  };

  // ---- prep ----
  f32_to_f16_k<<<(NN * 96 + 255) / 256, 256, 0, stream>>>(F(IN_H1), h1h, NN * 96);
  f32_to_f16_k<<<(NN * 96 + 255) / 256, 256, 0, stream>>>(F(IN_H2), h2h, NN * 96);

  const int pfb[2] = { PF1, PF2 };
  for (int f = 0; f < 2; ++f) {
    int b = pfb[f];
    packB(F(b + F_ESE_W1),  fp[f].w1p,  20, 128, 32);
    packB(F(b + F_GATE_W1), fp[f].gw1p, 192, 128, 192);
    packB(F(b + F_PAY_W1),  fp[f].pw1p, 96, 128, 96);
    packB(F(b + F_PAY_W2),  fp[f].pw2p, 128, 64, 128);
    fold_bw_k<<<1, 128, 0, stream>>>(F(b + F_ESE_W2), F(b + F_ESE_B2),
                                     F(b + F_BW), F(b + F_BB), fp[f].wv, fp[f].c0);
  }
  const int pgb[2] = { PG12, PG21 };
  for (int g = 0; g < 2; ++g) {
    int b = pgb[g];
    packB(F(b + G_WL),     gp[g].wlp,   96, 64, 96);
    packB(F(b + G_WR),     gp[g].wrp,   96, 64, 96);
    packB(F(b + G_WRES),   gp[g].wresp, 96, 64, 96);
    packB(F(b + G_FFN_W1), gp[g].f1p,   64, 128, 64);
    packB(F(b + G_FFN_W2), gp[g].f2p,   128, 64, 128);
  }
  const int pnb[2] = { PN1, PN2 };
  for (int t = 0; t < 2; ++t) {
    int b = pnb[t];
    packB(F(b + ND_WIH), np[t].wihp, 128, 288, 128);
    packB(F(b + ND_WHH), np[t].whhp, 96, 288, 96);
  }

  fill(msg1, 0.f, NN * 64);
  fill(msg2, 0.f, NN * 64);

  // ---- StaticDynamicEdgeMP (flow1 -> msg1, flow2 -> msg2) ----
  for (int f = 0; f < 2; ++f) {
    int b = pfb[f];
    sd_edge_k<<<EFn / 64, 128, 0, stream>>>(
        I(f == 0 ? IN_EIF1 : IN_EIF2), F(f == 0 ? IN_EA1 : IN_EA2),
        F(f == 0 ? IN_XS1 : IN_XS2), (f == 0 ? h1h : h2h),
        fp[f].w1p, F(b + F_ESE_B1), fp[f].wv, fp[f].c0,
        fp[f].gw1p, F(b + F_GATE_B1), F(b + F_GATE_W2), F(b + F_GATE_B2),
        fp[f].pw1p, F(b + F_PAY_B1), fp[f].pw2p, F(b + F_PAY_B2),
        (f == 0 ? msg1 : msg2), EFn);
  }

  // ---- GATv2 cross-type relations (buffers reused sequentially) ----
  auto run_gat = [&](int pb, GP& g, const _Float16* hsrc, const _Float16* hdst,
                     const int* ei, float* msg) {
    gemm(hsrc, 96, g.wlp, F(pb + G_BL), nullptr, xl, nullptr, NN, 96, 64, 0);
    gemm(hdst, 96, g.wrp, F(pb + G_BR), nullptr, xr, nullptr, NN, 96, 64, 0);
    fill(aout, 0.f, NN * 64);
    fill(mbuf, -__builtin_huge_valf(), NN * 4);
    fill(ssum, 0.f, NN * 4);
    int T = ECn * 4;
    gat_logits_k<<<(T + 255) / 256, 256, 0, stream>>>(ei, xl, xr, F(pb + G_ATT), est, mbuf, ECn);
    gat_exp_k<<<(T + 255) / 256, 256, 0, stream>>>(ei, est, mbuf, ssum, ECn);
    gat_scatter_k<<<(T + 255) / 256, 256, 0, stream>>>(ei, est, ssum, xl, aout, ECn);
    // residual: attin = aout + h_dst @ wres + bias (f16 for FFN input)
    gemm(hdst, 96, g.wresp, F(pb + G_BIAS), aout, nullptr, attin, NN, 96, 64, 0);
    gemm(attin, 64, g.f1p, F(pb + G_FFN_B1), nullptr, nullptr, ffnh, NN, 64, 128, 1);
    gemm(ffnh, 128, g.f2p, F(pb + G_FFN_B2), msg, msg, nullptr, NN, 128, 64, 0);
  };
  run_gat(PG21, gp[1], h2h, h1h, I(IN_EI21), msg1);  // twoD -> oneD
  run_gat(PG12, gp[0], h1h, h2h, I(IN_EI12), msg2);  // oneD -> twoD

  // ---- Node updates (GRU + LayerNorms), gi/gh/xcat reused ----
  float* out = (float*)d_out;
  auto run_node = [&](int pb, NP& n, const float* xdyn, const float* msg,
                      const _Float16* hh, const float* hf, float* outp) {
    int T = NN * 32;
    node_pre_k<<<(T + 255) / 256, 256, 0, stream>>>(
        xdyn, F(pb + ND_DYN_W), F(pb + ND_DYN_B),
        F(pb + ND_DLN_G), F(pb + ND_DLN_B), msg,
        F(pb + ND_MLN_G), F(pb + ND_MLN_B), xcat, NN);
    gemm(xcat, 128, n.wihp, F(pb + ND_BIH), nullptr, gi, nullptr, NN, 128, 288, 0);
    gemm(hh, 96, n.whhp, F(pb + ND_BHH), nullptr, gh, nullptr, NN, 96, 288, 0);
    gru_ln_k<<<(T + 255) / 256, 256, 0, stream>>>(
        gi, gh, hf, F(pb + ND_HLN_G), F(pb + ND_HLN_B), outp, NN);
  };
  run_node(PN1, np[0], F(IN_XD1), msg1, h1h, F(IN_H1), out);
  run_node(PN2, np[1], F(IN_XD2), msg2, h2h, F(IN_H2), out + (size_t)NN * 96);
}
